// VectorQuantizer_1563368096095
// MI455X (gfx1250) — compile-verified, matrix operand order checked by reading, not run on hardware
//
#include <hip/hip_runtime.h>
#include <stdint.h>

// ---------------------------------------------------------------------------
// VQ-VAE vector quantizer for MI455X (gfx1250, wave32, WMMA).
//
// x:           [B=32, C=64, H=64, W=64] fp32    (d_in[0])
// embeddings:  [C=64, K=512] fp32               (d_in[1])
// d_out (float, concatenated):
//   [0]                      loss = 1.25 * mean((q - z)^2)
//   [1 .. 1+8388608)         quantized, NCHW
//   [8388609]                perplexity
//   [8388610 .. +131072)     argmin indices (as float)
//
// dist_k (mod row-constant ||z||^2) = ||e_k||^2 - 2 z.e_k, computed as
// WMMA(A = -2z bf16, B = e bf16, C = ||e||^2 broadcast) with fp32 accum.
// B fragments are pre-swizzled to WMMA layout and staged in LDS (64 KB)
// once per 8-wave workgroup to keep codebook traffic off HBM/L2.
// ---------------------------------------------------------------------------

#define Bn   32
#define Cn   64
#define Hn   64
#define Wn   64
#define Kn   512
#define Nn   (Bn * Hn * Wn)          // 131072 pixels
#define HWn  (Hn * Wn)               // 4096 (channel stride in x)
#define NTILE (Nn / 16)              // 8192 pixel tiles
#define KTILES (Kn / 16)             // 32 code tiles
#define WAVES_PER_BLOCK 8

// workspace layout (dwords)
#define WS_B_DW     0                 // 32 tiles * 2 frags * 32 lanes * 8 dw = 16384
#define WS_ENORM_DW 16384             // 512 floats
#define WS_HIST_DW  16896             // 512 floats (zeroed each call)
#define WS_SUM_DW   17408             // 1 float    (zeroed each call)

typedef __attribute__((ext_vector_type(16))) __bf16 v16bf;
typedef __attribute__((ext_vector_type(8)))  float  v8f;

union BFrag { uint32_t u[8]; v16bf v; };

__device__ __forceinline__ uint32_t f2bf(float x) {
  uint32_t u = __float_as_uint(x);
  // round-to-nearest-even truncation to bf16
  uint32_t r = u + 0x7FFFu + ((u >> 16) & 1u);
  return r >> 16;
}

// ---------------------------------------------------------------------------
// Prep: bf16-swizzle embeddings into WMMA B-fragment layout + code norms.
// B fragment (32x16 bf16, K-dim = channels): lane L (col n = L&15, half =
// L>>4), VGPR v holds channels c_local = (v>>2)*16 + half*8 + (v&3)*2 (+1).
// Stored as wsB[((t*2+f)*32 + L)*8 + v] so waves read 2x b128 per fragment.
// ---------------------------------------------------------------------------
__global__ __launch_bounds__(256) void vq_prep(const float* __restrict__ emb,
                                               uint32_t* __restrict__ wsB,
                                               float* __restrict__ enorm) {
  int tid = blockIdx.x * blockDim.x + threadIdx.x;   // 16384 threads
  int v = tid & 7;
  int L = (tid >> 3) & 31;
  int f = (tid >> 8) & 1;
  int t = tid >> 9;
  int col  = L & 15;
  int hlf  = L >> 4;
  int k = t * 16 + col;
  int c = f * 32 + ((v >> 2) * 16) + hlf * 8 + ((v & 3) * 2);
  float e0 = emb[c * Kn + k];
  float e1 = emb[(c + 1) * Kn + k];
  wsB[tid] = f2bf(e0) | (f2bf(e1) << 16);

  if (tid < Kn) {
    float s = 0.f;
    #pragma unroll 8
    for (int cc = 0; cc < Cn; ++cc) {
      float e = emb[cc * Kn + tid];
      s += e * e;
    }
    enorm[tid] = s;
  }
}

// ---------------------------------------------------------------------------
// Main: 8 waves / block; one wave per 16-pixel tile; B table shared in LDS.
// ---------------------------------------------------------------------------
__global__ __launch_bounds__(256) void vq_main(const float* __restrict__ x,
                                               const float* __restrict__ emb,
                                               const uint32_t* __restrict__ wsB,
                                               const float* __restrict__ enorm,
                                               float* __restrict__ hist,
                                               float* __restrict__ sumsq,
                                               float* __restrict__ qout,
                                               float* __restrict__ idxout) {
  __shared__ uint32_t sB[16384];               // 64 KB staged B fragments

  // ---- cooperative stage of the swizzled codebook into LDS ----
  {
    const uint4* g = reinterpret_cast<const uint4*>(wsB);
    uint4* s = reinterpret_cast<uint4*>(sB);
    #pragma unroll 4
    for (int i = threadIdx.x; i < 4096; i += 256) s[i] = g[i];
  }
  __syncthreads();

  const int wave = threadIdx.x >> 5;
  const int lane = threadIdx.x & 31;
  const int T    = blockIdx.x * WAVES_PER_BLOCK + wave;
  const int n0   = T * 16;
  const int b    = n0 >> 12;          // / (H*W)
  const int h    = (n0 >> 6) & 63;
  const int w0   = n0 & 63;           // 16 | 64 -> tile stays within one row
  const int m    = lane & 15;         // pixel within tile (A/D row id base)
  const int hlf  = lane >> 4;
  const int col  = lane & 15;         // code column within a 16-wide tile

  // ---- build A fragments: a[m][c] = -2 * z[m][c], bf16 packed pairs ----
  BFrag A0, A1;
  #pragma unroll
  for (int f = 0; f < 2; ++f) {
    #pragma unroll
    for (int v = 0; v < 8; ++v) {
      int c = f * 32 + ((v >> 2) * 16) + hlf * 8 + ((v & 3) * 2);
      const float* px = x + (((size_t)(b * Cn + c) * Hn + h) * Wn + w0 + m);
      float z0 = px[0];
      float z1 = px[HWn];            // channel c+1
      uint32_t p = f2bf(-2.f * z0) | (f2bf(-2.f * z1) << 16);
      if (f == 0) A0.u[v] = p; else A1.u[v] = p;
    }
  }

  float minv[8];
  int   mink[8];
  #pragma unroll
  for (int r = 0; r < 8; ++r) { minv[r] = 3.4e38f; mink[r] = 0; }

  const uint4* sB4 = reinterpret_cast<const uint4*>(sB);

  #pragma unroll 2
  for (int t = 0; t < KTILES; ++t) {
    // B fragments from LDS: two b128 reads per lane per fragment
    BFrag B0, B1;
    {
      const uint4* p0 = sB4 + ((t * 2 + 0) * 32 + lane) * 2;
      const uint4* p1 = sB4 + ((t * 2 + 1) * 32 + lane) * 2;
      uint4 a = p0[0], bq = p0[1];
      B0.u[0]=a.x; B0.u[1]=a.y; B0.u[2]=a.z; B0.u[3]=a.w;
      B0.u[4]=bq.x; B0.u[5]=bq.y; B0.u[6]=bq.z; B0.u[7]=bq.w;
      a = p1[0]; bq = p1[1];
      B1.u[0]=a.x; B1.u[1]=a.y; B1.u[2]=a.z; B1.u[3]=a.w;
      B1.u[4]=bq.x; B1.u[5]=bq.y; B1.u[6]=bq.z; B1.u[7]=bq.w;
    }

    // fold ||e_k||^2 into the accumulator init (one value per lane-column,
    // broadcast across the 8 row registers of the C/D layout)
    float en = enorm[t * 16 + col];
    v8f d = {en, en, en, en, en, en, en, en};
    d = __builtin_amdgcn_wmma_f32_16x16x32_bf16(false, A0.v, false, B0.v,
                                                (short)0, d, false, false);
    d = __builtin_amdgcn_wmma_f32_16x16x32_bf16(false, A1.v, false, B1.v,
                                                (short)0, d, false, false);

    #pragma unroll
    for (int r = 0; r < 8; ++r) {
      float cand = d[r];
      if (cand < minv[r]) { minv[r] = cand; mink[r] = t; }
    }
  }

  // full code index per lane, then 16-lane butterfly argmin (ties -> low k)
  int kf[8];
  #pragma unroll
  for (int r = 0; r < 8; ++r) kf[r] = mink[r] * 16 + col;
  #pragma unroll
  for (int mask = 1; mask <= 8; mask <<= 1) {
    #pragma unroll
    for (int r = 0; r < 8; ++r) {
      float ov = __shfl_xor(minv[r], mask, 32);
      int   ok = __shfl_xor(kf[r],   mask, 32);
      if (ov < minv[r] || (ov == minv[r] && ok < kf[r])) {
        minv[r] = ov; kf[r] = ok;
      }
    }
  }

  // Recover this lane's pixel winner km = winner(row m) without LDS:
  // after the butterfly, kf[r] (all lanes of half hlf) = winner(row r+8*hlf).
  int km = 0;
  #pragma unroll
  for (int r = 0; r < 8; ++r) {
    int other   = __shfl_xor(kf[r], 16, 32);
    int lowRow  = hlf ? other : kf[r];     // winner(row r)
    int highRow = hlf ? kf[r] : other;     // winner(row r+8)
    int sel = (m & 8) ? highRow : lowRow;
    if ((m & 7) == r) km = sel;
  }

  // indices output: lanes 0..15 hold winners of rows 0..15 in km
  if (lane < 16) idxout[n0 + lane] = (float)km;

  // gather quantized, write NCHW output, accumulate sum((q - z)^2)
  float ssum = 0.f;
  #pragma unroll 4
  for (int it = 0; it < 32; ++it) {
    int c = 2 * it + hlf;
    size_t off = ((size_t)(b * Cn + c) * Hn + h) * Wn + w0 + m;
    float q = emb[c * Kn + km];
    float z = x[off];
    float dd = q - z;
    ssum += dd * dd;
    qout[off] = q;
  }
  #pragma unroll
  for (int mask = 16; mask; mask >>= 1) ssum += __shfl_xor(ssum, mask, 32);
  if (lane == 0) atomicAdd(sumsq, ssum);
  if (lane < 16) atomicAdd(&hist[km], 1.0f);
}

// ---------------------------------------------------------------------------
// Finalize scalars.
// ---------------------------------------------------------------------------
__global__ __launch_bounds__(32) void vq_finalize(const float* __restrict__ hist,
                                                  const float* __restrict__ sumsq,
                                                  float* __restrict__ out) {
  int lane = threadIdx.x;
  float s = 0.f;
  for (int k = lane; k < Kn; k += 32) {
    float p = hist[k] * (1.f / (float)Nn);
    s += p * logf(p + 1e-10f);
  }
  #pragma unroll
  for (int mask = 16; mask; mask >>= 1) s += __shfl_xor(s, mask, 32);
  if (lane == 0) {
    out[1 + (size_t)Nn * Cn] = expf(-s);                       // perplexity
    out[0] = 1.25f * sumsq[0] / ((float)Nn * (float)Cn);       // loss
  }
}

extern "C" void kernel_launch(void* const* d_in, const int* in_sizes, int n_in,
                              void* d_out, int out_size, void* d_ws, size_t ws_size,
                              hipStream_t stream) {
  (void)in_sizes; (void)n_in; (void)out_size; (void)ws_size;
  const float* x   = (const float*)d_in[0];
  const float* emb = (const float*)d_in[1];
  float* out = (float*)d_out;

  uint32_t* ws    = (uint32_t*)d_ws;
  uint32_t* wsB   = ws + WS_B_DW;
  float*    enorm = (float*)(ws + WS_ENORM_DW);
  float*    hist  = (float*)(ws + WS_HIST_DW);
  float*    sumsq = (float*)(ws + WS_SUM_DW);

  // zero histogram + loss accumulator (capture-safe, deterministic per call)
  hipMemsetAsync((char*)d_ws + WS_HIST_DW * 4, 0, (Kn + 1) * 4, stream);

  vq_prep<<<64, 256, 0, stream>>>(emb, wsB, enorm);

  float* qout   = out + 1;
  float* idxout = out + 2 + (size_t)Nn * Cn;
  vq_main<<<NTILE / WAVES_PER_BLOCK, 256, 0, stream>>>(x, emb, wsB, enorm,
                                                       hist, sumsq, qout, idxout);

  vq_finalize<<<1, 32, 0, stream>>>(hist, sumsq, out);
}
